// PointNetpp_60490319396909
// MI455X (gfx1250) — compile-verified, matrix operand order checked by reading, not run on hardware
//
#include <hip/hip_runtime.h>
#include <hip/hip_bf16.h>

typedef __attribute__((ext_vector_type(16))) __bf16 v16bf;
typedef __attribute__((ext_vector_type(8)))  float  v8f;

#define BATCH 4

__device__ inline unsigned short f2bf_bits(float f) {
    union { __bf16 b; unsigned short u; } cv;
    cv.b = (__bf16)f;
    return cv.u;
}

// ---------------------------------------------------------------------------
// Farthest point sampling: one block per batch, dist[] resident in LDS.
// Matches reference: idx[0]=0, then repeatedly argmax of running min-dist.
// ---------------------------------------------------------------------------
__global__ __launch_bounds__(256) void fps_kernel(const float* __restrict__ xyz,
                                                  int N, int npoint,
                                                  int* __restrict__ cidx) {
    __shared__ float dist[8192];     // N <= 8192
    __shared__ float rmax[256];
    __shared__ int   rarg[256];
    const int b = blockIdx.x, tid = threadIdx.x;
    const float* p = xyz + (long)b * N * 3;
    for (int i = tid; i < N; i += 256) dist[i] = 1e10f;
    __syncthreads();
    int far = 0;
    for (int it = 0; it < npoint; ++it) {
        if (tid == 0) cidx[b * npoint + it] = far;
        const float cx = p[far*3+0], cy = p[far*3+1], cz = p[far*3+2];
        float best = -1.0f; int bi = 0;
        for (int i = tid; i < N; i += 256) {
            float dx = p[i*3+0]-cx, dy = p[i*3+1]-cy, dz = p[i*3+2]-cz;
            float d  = dx*dx + dy*dy + dz*dz;
            float nd = fminf(dist[i], d);
            dist[i]  = nd;
            if (nd > best) { best = nd; bi = i; }
        }
        rmax[tid] = best; rarg[tid] = bi;
        __syncthreads();
        for (int s = 128; s > 0; s >>= 1) {
            if (tid < s) {
                if (rmax[tid+s] > rmax[tid] ||
                    (rmax[tid+s] == rmax[tid] && rarg[tid+s] < rarg[tid])) {
                    rmax[tid] = rmax[tid+s]; rarg[tid] = rarg[tid+s];
                }
            }
            __syncthreads();
        }
        far = rarg[0];
        __syncthreads();
    }
}

__global__ void gather_cent_kernel(const float* __restrict__ xyz,
                                   const int* __restrict__ cidx,
                                   float* __restrict__ cent, int N, int S) {
    int t = blockIdx.x * 256 + threadIdx.x;
    if (t >= BATCH * S) return;
    int b = t / S;
    int c = cidx[t];
    const float* src = xyz + ((long)b * N + c) * 3;
    cent[t*3+0] = src[0];
    cent[t*3+1] = src[1];
    cent[t*3+2] = src[2];
}

// Reference ball query: first K in-radius indices (ascending), pad with first.
__global__ void ball_query_kernel(const float* __restrict__ xyz,
                                  const float* __restrict__ cent,
                                  int* __restrict__ gidx,
                                  int N, int S, int K, float r2) {
    int t = blockIdx.x * 256 + threadIdx.x;
    if (t >= BATCH * S) return;
    int b = t / S;
    const float* p = xyz + (long)b * N * 3;
    const float cx = cent[t*3+0], cy = cent[t*3+1], cz = cent[t*3+2];
    int* out = gidx + (long)t * K;
    int cnt = 0;
    for (int n = 0; n < N && cnt < K; ++n) {
        float dx = p[n*3+0]-cx, dy = p[n*3+1]-cy, dz = p[n*3+2]-cz;
        if (dx*dx + dy*dy + dz*dz <= r2) out[cnt++] = n;
    }
    int fill = (cnt > 0) ? out[0] : 0;
    for (; cnt < K; ++cnt) out[cnt] = fill;
}

// Gather grouped features -> A matrix [M, Kpad] bf16, zero-padded channels.
__global__ void group_pad_cvt_kernel(const float* __restrict__ feats,
                                     const int* __restrict__ gidx,
                                     __bf16* __restrict__ A,
                                     long M, int Npts, int Cin, int Kpad) {
    long e = (long)blockIdx.x * 256 + threadIdx.x;
    if (e >= M * (long)Kpad) return;
    long r = e / Kpad;
    int  c = (int)(e - r * Kpad);
    int  b = (int)(r / (M / BATCH));
    int  idx = gidx[r];                      // gidx flat layout == row index
    float v = (c < Cin) ? feats[((long)b * Npts + idx) * Cin + c] : 0.0f;
    A[e] = (__bf16)v;
}

// Pack weights w[K,N] f32 into WMMA B-fragment layout (bf16 pairs as u32):
//   pk[((nt*KT + kt)*32 + lane)*8 + d] covers n = nt*16 + lane%16,
//   k0 = kt*32 + (d/4)*16 + (lane/16)*8 + (d%4)*2  (pair k0, k0+1)
__global__ void pack_weights_kernel(const float* __restrict__ w,
                                    unsigned int* __restrict__ pk,
                                    int K, int Kpad, int N) {
    long t = (long)blockIdx.x * 256 + threadIdx.x;
    int KT = Kpad >> 5;
    long total = (long)(N >> 4) * KT * 256;
    if (t >= total) return;
    int  d    = (int)(t & 7);
    int  lane = (int)((t >> 3) & 31);
    long g2   = t >> 8;
    int  kt   = (int)(g2 % KT);
    int  nt   = (int)(g2 / KT);
    int  n    = nt * 16 + (lane & 15);
    int  k0   = kt * 32 + ((d >> 2) << 4) + ((lane >> 4) << 3) + ((d & 3) << 1);
    unsigned short lo = f2bf_bits((k0     < K) ? w[(long)k0       * N + n] : 0.0f);
    unsigned short hi = f2bf_bits((k0 + 1 < K) ? w[(long)(k0 + 1) * N + n] : 0.0f);
    pk[t] = (unsigned int)lo | ((unsigned int)hi << 16);
}

// ---------------------------------------------------------------------------
// WMMA GEMM: X[M,N] = A[M,Kpad](bf16) * W[Kpad,N] + bias.
// Block = 8 waves arranged 4(M) x 2(N-groups); each wave computes a 16x64
// strip: one A fragment feeds 4 B fragments / 4 accumulators per K-step.
// ---------------------------------------------------------------------------
__global__ __launch_bounds__(256) void wmma_gemm_bf16(
        const __bf16* __restrict__ A, const unsigned int* __restrict__ pkB,
        const float* __restrict__ bias, float* __restrict__ X,
        int M, int Kpad, int N) {
    const int lane  = threadIdx.x & 31;
    const int wave  = threadIdx.x >> 5;
    const int mi    = wave & 3;
    const int ni    = wave >> 2;
    const int tileM = blockIdx.x * 4 + mi;
    const int nt0   = (blockIdx.y * 2 + ni) * 4;   // first of 4 N-tiles
    if (nt0 * 16 >= N) return;
    const int  KT   = Kpad >> 5;
    const int  mloc = lane & 15;
    const int  half = lane >> 4;
    const long arow = (long)(tileM * 16 + mloc) * Kpad;

    v8f acc0 = {0.f,0.f,0.f,0.f,0.f,0.f,0.f,0.f};
    v8f acc1 = acc0, acc2 = acc0, acc3 = acc0;
    union Frag { v16bf v; unsigned int u[8]; } a, b0, b1, b2, b3;
    const long bstep = (long)KT * 256;              // u32s per N-tile
    const unsigned int* bp = pkB + (long)nt0 * bstep + lane * 8;

    for (int kt = 0; kt < KT; ++kt) {
        const unsigned int* ap = (const unsigned int*)(A + arow + (long)kt * 32);
        const unsigned int* bk = bp + (long)kt * 256;
        __builtin_prefetch(bk + 256, 0, 3);
#pragma unroll
        for (int v = 0; v < 8; ++v) {
            int k0 = ((v >> 2) << 4) + (half << 3) + ((v & 3) << 1);
            a.u[v] = ap[k0 >> 1];
        }
#pragma unroll
        for (int v = 0; v < 8; ++v) b0.u[v] = bk[v];
#pragma unroll
        for (int v = 0; v < 8; ++v) b1.u[v] = bk[bstep + v];
#pragma unroll
        for (int v = 0; v < 8; ++v) b2.u[v] = bk[2 * bstep + v];
#pragma unroll
        for (int v = 0; v < 8; ++v) b3.u[v] = bk[3 * bstep + v];
        acc0 = __builtin_amdgcn_wmma_f32_16x16x32_bf16(
                false, a.v, false, b0.v, (short)0, acc0, false, false);
        acc1 = __builtin_amdgcn_wmma_f32_16x16x32_bf16(
                false, a.v, false, b1.v, (short)0, acc1, false, false);
        acc2 = __builtin_amdgcn_wmma_f32_16x16x32_bf16(
                false, a.v, false, b2.v, (short)0, acc2, false, false);
        acc3 = __builtin_amdgcn_wmma_f32_16x16x32_bf16(
                false, a.v, false, b3.v, (short)0, acc3, false, false);
    }

    const long rbase = (long)tileM * 16 + (half << 3);
#pragma unroll
    for (int j = 0; j < 4; ++j) {
        const int   ncol = (nt0 + j) * 16 + mloc;
        const float bv   = bias[ncol];
        const v8f   accv = (j == 0) ? acc0 : (j == 1) ? acc1 : (j == 2) ? acc2 : acc3;
#pragma unroll
        for (int v = 0; v < 8; ++v) {
            X[(rbase + v) * N + ncol] = accv[v] + bv;
        }
    }
}

// Per-channel sum / sumsq over M rows (coalesced: consecutive threads = cols).
__global__ void bn_stats_kernel(const float* __restrict__ X,
                                float* __restrict__ stats, long M, int N) {
    int g = blockIdx.x * 256 + threadIdx.x;
    if (g >= N * 1024) return;
    int n = g % N;
    int s = g / N;
    float sum = 0.0f, sq = 0.0f;
    for (long r = s; r < M; r += 1024) {
        float v = X[r * N + n];
        sum += v; sq += v * v;
    }
    atomicAdd(&stats[n], sum);
    atomicAdd(&stats[N + n], sq);
}

// BN + ReLU, convert to bf16 (next GEMM's A matrix).
__global__ void bn_relu_cvt_kernel(const float* __restrict__ X,
                                   const float* __restrict__ stats,
                                   const float* __restrict__ gam,
                                   const float* __restrict__ bet,
                                   __bf16* __restrict__ H, long M, int N) {
    long e = (long)blockIdx.x * 256 + threadIdx.x;
    if (e >= M * (long)N) return;
    int n = (int)(e % N);
    float mean = stats[n] / (float)M;
    float var  = stats[N + n] / (float)M - mean * mean;
    float sc   = gam[n] * rsqrtf(var + 1e-5f);
    float v    = (X[e] - mean) * sc + bet[n];
    H[e] = (__bf16)fmaxf(v, 0.0f);
}

// BN + ReLU + max over the K (nsample) axis -> layer output feats [BS, N].
__global__ void bn_relu_maxpool_kernel(const float* __restrict__ X,
                                       const float* __restrict__ stats,
                                       const float* __restrict__ gam,
                                       const float* __restrict__ bet,
                                       float* __restrict__ out,
                                       int BS, int K, int N, long M) {
    int t = blockIdx.x * 256 + threadIdx.x;
    if (t >= BS * N) return;
    int n  = t % N;
    int bs = t / N;
    float mean = stats[n] / (float)M;
    float var  = stats[N + n] / (float)M - mean * mean;
    float sc   = gam[n] * rsqrtf(var + 1e-5f);
    float bb   = bet[n];
    float m = 0.0f;                       // relu output >= 0
    for (int k = 0; k < K; ++k) {
        float v = (X[((long)bs * K + k) * N + n] - mean) * sc + bb;
        m = fmaxf(m, fmaxf(v, 0.0f));
    }
    out[(long)bs * N + n] = m;
}

__global__ void smax_kernel(const float* __restrict__ f, float* __restrict__ g,
                            int S, int C) {
    int t = blockIdx.x * 256 + threadIdx.x;
    if (t >= BATCH * C) return;
    int c = t % C, b = t / C;
    float m = -1e30f;
    for (int s = 0; s < S; ++s) m = fmaxf(m, f[((long)b * S + s) * C + c]);
    g[t] = m;
}

__global__ void fc_kernel(const float* __restrict__ in, const float* __restrict__ w,
                          const float* __restrict__ b, float* __restrict__ out,
                          int Bn, int Cin, int Cout, int relu) {
    int t = blockIdx.x * 256 + threadIdx.x;
    if (t >= Bn * Cout) return;
    int o = t % Cout, bb = t / Cout;
    float acc = b[o];
    for (int i = 0; i < Cin; ++i) acc += in[bb * Cin + i] * w[(long)i * Cout + o];
    out[t] = relu ? fmaxf(acc, 0.0f) : acc;
}

// ---------------------------------------------------------------------------
extern "C" void kernel_launch(void* const* d_in, const int* in_sizes, int n_in,
                              void* d_out, int out_size, void* d_ws, size_t ws_size,
                              hipStream_t stream) {
    const float* x = (const float*)d_in[0];
    struct LW { const float *w1,*b1,*g1,*be1,*w2,*b2,*g2,*be2; } lw[3];
    for (int l = 0; l < 3; ++l) {
        int k = 1 + l * 8;
        lw[l].w1  = (const float*)d_in[k+0]; lw[l].b1  = (const float*)d_in[k+1];
        lw[l].g1  = (const float*)d_in[k+2]; lw[l].be1 = (const float*)d_in[k+3];
        lw[l].w2  = (const float*)d_in[k+4]; lw[l].b2  = (const float*)d_in[k+5];
        lw[l].g2  = (const float*)d_in[k+6]; lw[l].be2 = (const float*)d_in[k+7];
    }
    const float* fc1w = (const float*)d_in[25];
    const float* fc1b = (const float*)d_in[26];
    const float* fc2w = (const float*)d_in[27];
    const float* fc2b = (const float*)d_in[28];

    // --- bump allocator over d_ws ---
    size_t off = 0;
    char* base = (char*)d_ws;
    auto alloc = [&](size_t bytes) -> void* {
        void* p = base + off;
        off = (off + bytes + 255) & ~(size_t)255;
        return p;
    };
    int*   cidx   = (int*)  alloc((size_t)BATCH * 4096 * 4);
    int*   gidx   = (int*)  alloc((size_t)BATCH * 4096 * 32 * 4);   // max B*S*K
    float* cent1  = (float*)alloc((size_t)BATCH * 4096 * 3 * 4);
    float* cent2  = (float*)alloc((size_t)BATCH * 1024 * 3 * 4);
    float* cent3  = (float*)alloc((size_t)BATCH *  256 * 3 * 4);
    float* feats1 = (float*)alloc((size_t)BATCH * 4096 * 128 * 4);
    float* feats2 = (float*)alloc((size_t)BATCH * 1024 * 256 * 4);
    float* feats3 = (float*)alloc((size_t)BATCH *  256 * 512 * 4);
    float* stats  = (float*)alloc((size_t)2 * 512 * 4);
    unsigned int* pk1 = (unsigned int*)alloc((size_t)2 * 256 * 256);
    unsigned int* pk2 = (unsigned int*)alloc((size_t)2 * 256 * 512);
    __bf16* bufA = (__bf16*)alloc((size_t)524288 * 64 * 2);   // grouped, then H1 (67 MB)
    float*  bufX = (float*) alloc((size_t)524288 * 128 * 4);  // X1, then X2 (268 MB)
    float*  gbuf = (float*) alloc((size_t)BATCH * 512 * 4);
    float*  hbuf = (float*) alloc((size_t)BATCH * 128 * 4);

    const int Ns[3]  = {8192, 4096, 1024};
    const int Ss[3]  = {4096, 1024,  256};
    const int Ksm[3] = {  32,   64,  128};
    const int Cin[3] = {   3,  128,  256};
    const int Kp[3]  = {  32,  128,  256};
    const int C1[3]  = {  64,  128,  256};
    const int C2[3]  = { 128,  256,  512};
    const float r2 = 0.25f;

    const float* xyz = x;
    const float* fin = x;
    float* cents[3]  = {cent1, cent2, cent3};
    float* fouts[3]  = {feats1, feats2, feats3};

    for (int l = 0; l < 3; ++l) {
        const int N = Ns[l], S = Ss[l], K = Ksm[l];
        const long M = (long)BATCH * S * K;

        fps_kernel<<<BATCH, 256, 0, stream>>>(xyz, N, S, cidx);
        gather_cent_kernel<<<(BATCH*S + 255)/256, 256, 0, stream>>>(xyz, cidx, cents[l], N, S);
        ball_query_kernel<<<(BATCH*S + 255)/256, 256, 0, stream>>>(xyz, cents[l], gidx, N, S, K, r2);
        group_pad_cvt_kernel<<<(unsigned)((M*Kp[l] + 255)/256), 256, 0, stream>>>(
            fin, gidx, bufA, M, N, Cin[l], Kp[l]);

        // GEMM1 + BN stats + BN/ReLU -> bf16 H
        {
            long npk = (long)(C1[l]/16) * (Kp[l]/32) * 256;
            pack_weights_kernel<<<(unsigned)((npk + 255)/256), 256, 0, stream>>>(
                lw[l].w1, pk1, Cin[l], Kp[l], C1[l]);
            dim3 g((unsigned)(M/64), (unsigned)((C1[l] + 127)/128));
            wmma_gemm_bf16<<<g, 256, 0, stream>>>(bufA, pk1, lw[l].b1, bufX,
                                                  (int)M, Kp[l], C1[l]);
            hipMemsetAsync(stats, 0, (size_t)2*C1[l]*4, stream);
            bn_stats_kernel<<<(unsigned)(C1[l]*4), 256, 0, stream>>>(bufX, stats, M, C1[l]);
            bn_relu_cvt_kernel<<<(unsigned)((M*C1[l] + 255)/256), 256, 0, stream>>>(
                bufX, stats, lw[l].g1, lw[l].be1, bufA, M, C1[l]);
        }
        // GEMM2 + BN stats + BN/ReLU + maxpool over K -> feats
        {
            long npk = (long)(C2[l]/16) * (C1[l]/32) * 256;
            pack_weights_kernel<<<(unsigned)((npk + 255)/256), 256, 0, stream>>>(
                lw[l].w2, pk2, C1[l], C1[l], C2[l]);
            dim3 g((unsigned)(M/64), (unsigned)((C2[l] + 127)/128));
            wmma_gemm_bf16<<<g, 256, 0, stream>>>(bufA, pk2, lw[l].b2, bufX,
                                                  (int)M, C1[l], C2[l]);
            hipMemsetAsync(stats, 0, (size_t)2*C2[l]*4, stream);
            bn_stats_kernel<<<(unsigned)(C2[l]*4), 256, 0, stream>>>(bufX, stats, M, C2[l]);
            bn_relu_maxpool_kernel<<<(unsigned)((BATCH*S*C2[l] + 255)/256), 256, 0, stream>>>(
                bufX, stats, lw[l].g2, lw[l].be2, fouts[l], BATCH*S, K, C2[l], M);
        }
        xyz = cents[l];
        fin = fouts[l];
    }

    // Head: global max over S, then FC -> ReLU -> FC.
    smax_kernel<<<(BATCH*512 + 255)/256, 256, 0, stream>>>(feats3, gbuf, 256, 512);
    fc_kernel<<<(BATCH*128 + 255)/256, 256, 0, stream>>>(gbuf, fc1w, fc1b, hbuf, BATCH, 512, 128, 1);
    fc_kernel<<<1, 256, 0, stream>>>(hbuf, fc2w, fc2b, (float*)d_out, BATCH, 128, 33, 0);
}
